// StageAttentionwCAM_43739946942680
// MI455X (gfx1250) — matrix operand order repaired and checked by reading, not compiled, not run on hardware
//
#include <hip/hip_runtime.h>

#define HWp 36864
#define Wp  192
#define Hp  192
#define EPSf 1e-5f

typedef __attribute__((ext_vector_type(16))) _Float16 v16h;
typedef __attribute__((ext_vector_type(8)))  float    v8f;

__device__ __forceinline__ v8f wmma16(v16h a, v16h b, v8f c) {
  return __builtin_amdgcn_wmma_f32_16x16x32_f16(false, a, false, b, (short)0, c, false, false);
}
// A-matrix (16x32 f16) element->K mapping: contiguous k = hh*16 + e (matches B fills)
__device__ __forceinline__ int kA16(int e, int hh) { return hh * 16 + e; }

__device__ __forceinline__ float wred_sum(float v) {
  for (int o = 16; o; o >>= 1) v += __shfl_xor(v, o, 32);
  return v;
}
__device__ __forceinline__ float wred_max(float v) {
  for (int o = 16; o; o >>= 1) v = fmaxf(v, __shfl_xor(v, o, 32));
  return v;
}
__device__ __forceinline__ float sigmoidf(float x) { return 1.0f / (1.0f + __expf(-x)); }

__device__ __forceinline__ float pos_embed(int c, int h, int w) {
  int i; float pos;
  if (c < 32) { i = c >> 1; pos = (float)h; }
  else        { i = (c - 32) >> 1; pos = (float)w; }
  float dv = __expf(-9.210340371976184f * (float)i * (1.0f / 16.0f));
  float arg = pos * dv;
  return (c & 1) ? __cosf(arg) : __sinf(arg);
}

// ---------------- elementwise / pooling ----------------
__global__ void k_maxpool2(const float* __restrict__ src, float* __restrict__ dst) {
  int i = blockIdx.x * 256 + threadIdx.x;            // 9437184 total (exact)
  int w = i % Wp, h = (i / Wp) % Hp, bc = i / HWp;
  const float* s = src + (size_t)bc * 147456 + (size_t)(2 * h) * 384 + 2 * w;
  dst[i] = fmaxf(fmaxf(s[0], s[1]), fmaxf(s[384], s[385]));
}

__global__ void k_cvt_w(const float* __restrict__ w, _Float16* __restrict__ dst,
                        int rows, int cols) {
  int i = blockIdx.x * 256 + threadIdx.x;            // padRows*cols exact multiple of 256
  int r = i / cols;
  dst[i] = (r < rows) ? (_Float16)w[i] : (_Float16)0.0f;
}

// ---------------- WMMA 1x1 conv / channel GEMM ----------------
// dst[(b*Cout+mo)*HW + n] = sum_c A_b[mo*64+c] * src[(b*64+c)*HW + n]
template <int Cout, int Mtiles>
__global__ void k_conv1x1(const float* __restrict__ src, float* __restrict__ dst,
                          const _Float16* __restrict__ A, int aStride) {
  int gw = blockIdx.x * 4 + (threadIdx.x >> 5);
  int lane = threadIdx.x & 31;
  int npb = HWp / 16;
  int b = gw / npb, n0 = (gw % npb) * 16;
  int col = lane & 15, hh = lane >> 4, m = col;
  const float* xb = src + (size_t)b * 64 * HWp + n0 + col;
  v16h b0, b1;
#pragma unroll
  for (int e = 0; e < 16; e++) {
    int k = hh * 16 + e;
    b0[e] = (_Float16)xb[(size_t)k * HWp];
    b1[e] = (_Float16)xb[(size_t)(k + 32) * HWp];
  }
  const _Float16* Ab = A + (size_t)b * aStride;
#pragma unroll
  for (int mt = 0; mt < Mtiles; mt++) {
    v16h a0, a1;
#pragma unroll
    for (int e = 0; e < 16; e++) {
      int k = kA16(e, hh);
      a0[e] = Ab[(mt * 16 + m) * 64 + k];
      a1[e] = Ab[(mt * 16 + m) * 64 + 32 + k];
    }
    v8f c = {};
    c = wmma16(a0, b0, c);
    c = wmma16(a1, b1, c);
#pragma unroll
    for (int r = 0; r < 8; r++) {
      int mo = mt * 16 + hh * 8 + r;
      if (mo < Cout)   // compile-time foldable: full for Cout=64, hh==0 block for Cout=8
        dst[((size_t)b * Cout + mo) * HWp + n0 + col] = c[r];
    }
  }
}

// ---------------- BN over (b,h,w) ----------------
__global__ void k_bnstats(const float* __restrict__ x, float2* __restrict__ st, int C) {
  int ch = blockIdx.x, tid = threadIdx.x;
  float s = 0.f, s2 = 0.f;
  for (int b = 0; b < 4; b++) {
    const float* p = x + ((size_t)b * C + ch) * HWp;
    for (int i = tid; i < HWp; i += 256) { float v = p[i]; s += v; s2 += v * v; }
  }
  __shared__ float sh[256], sh2[256];
  sh[tid] = s; sh2[tid] = s2; __syncthreads();
  for (int o = 128; o; o >>= 1) {
    if (tid < o) { sh[tid] += sh[tid + o]; sh2[tid] += sh2[tid + o]; }
    __syncthreads();
  }
  if (tid == 0) {
    float n = 4.0f * HWp, mu = sh[0] / n, var = sh2[0] / n - mu * mu;
    st[ch] = make_float2(mu, rsqrtf(var + EPSf));
  }
}

__global__ void k_bnapply(float* __restrict__ x, const float2* __restrict__ st,
                          const float* __restrict__ g, const float* __restrict__ beta,
                          int C, int relu) {
  int i = blockIdx.x * 256 + threadIdx.x;            // exact grid
  int ch = (i / HWp) % C;
  float2 s = st[ch];
  float v = g[ch] * ((x[i] - s.x) * s.y) + beta[ch];
  if (relu) v = fmaxf(v, 0.f);
  x[i] = v;
}

// ---------------- SE ----------------
__global__ void k_semean(const float* __restrict__ x, float* __restrict__ seb) {
  int b = blockIdx.x >> 6, c = blockIdx.x & 63, tid = threadIdx.x;
  const float* p = x + ((size_t)b * 64 + c) * HWp;
  float s = 0.f;
  for (int i = tid; i < HWp; i += 256) s += p[i] + pos_embed(c, i / Wp, i % Wp);
  __shared__ float sh[256];
  sh[tid] = s; __syncthreads();
  for (int o = 128; o; o >>= 1) { if (tid < o) sh[tid] += sh[tid + o]; __syncthreads(); }
  if (tid == 0) seb[blockIdx.x] = sh[0] * (1.0f / (float)HWp);
}

__global__ void k_semlp(const float* __restrict__ seb, const float* __restrict__ w1,
                        const float* __restrict__ w2, float* __restrict__ scale) {
  int t = threadIdx.x;
  int b = t >> 6, c = t & 63;
  float h[4];
  for (int j = 0; j < 4; j++) {
    float a = 0.f;
    for (int cc = 0; cc < 64; cc++) a += seb[b * 64 + cc] * w1[j * 64 + cc];
    h[j] = fmaxf(a, 0.f);
  }
  float o = 0.f;
  for (int j = 0; j < 4; j++) o += h[j] * w2[c * 4 + j];
  scale[t] = sigmoidf(o);
}

__global__ void k_seapply(const float* __restrict__ x, const float* __restrict__ scale,
                          float* __restrict__ S) {
  int i = blockIdx.x * 256 + threadIdx.x;
  int hw = i % HWp, bc = i / HWp, c = bc & 63;
  S[i] = (x[i] + pos_embed(c, hw / Wp, hw % Wp)) * scale[bc];
}

// ---------------- attention logits (per batch) ----------------
// zero-padding of K (8 -> 32) done via unconditional clamped loads + v_cndmask
__global__ void k_eH(const float* __restrict__ q, const float* __restrict__ kk,
                     float* __restrict__ att, int b) {
  int gw = blockIdx.x * 4 + (threadIdx.x >> 5);
  int lane = threadIdx.x & 31;
  int iw = gw / 144, t = gw % 144, ih0 = (t / 12) * 16, j0 = (t % 12) * 16;
  int col = lane & 15, hh = lane >> 4, m = col;
  const float* qb = q + (size_t)b * 8 * HWp;
  const float* kb = kk + (size_t)b * 8 * HWp;
  v16h a, bb;
#pragma unroll
  for (int e = 0; e < 16; e++) {
    int kc = kA16(e, hh);
    float va = qb[(size_t)(kc & 7) * HWp + (ih0 + m) * Wp + iw];       // always in-bounds
    a[e] = (kc < 8) ? (_Float16)va : (_Float16)0.f;
    int kc2 = hh * 16 + e;
    float vb = kb[(size_t)(kc2 & 7) * HWp + (j0 + col) * Wp + iw];
    bb[e] = (kc2 < 8) ? (_Float16)vb : (_Float16)0.f;
  }
  v8f c = {};
  c = wmma16(a, bb, c);
  float ninf = __int_as_float(0xff800000);
#pragma unroll
  for (int r = 0; r < 8; r++) {
    int ih = ih0 + hh * 8 + r, j = j0 + col;
    att[((size_t)(ih * Wp + iw)) * 384 + j] = (ih == j) ? ninf : c[r];
  }
}

__global__ void k_eW(const float* __restrict__ q, const float* __restrict__ kk,
                     float* __restrict__ att, int b) {
  int gw = blockIdx.x * 4 + (threadIdx.x >> 5);
  int lane = threadIdx.x & 31;
  int ih = gw / 144, t = gw % 144, i0 = (t / 12) * 16, j0 = (t % 12) * 16;
  int col = lane & 15, hh = lane >> 4, m = col;
  const float* qb = q + (size_t)b * 8 * HWp;
  const float* kb = kk + (size_t)b * 8 * HWp;
  v16h a, bb;
#pragma unroll
  for (int e = 0; e < 16; e++) {
    int kc = kA16(e, hh);
    float va = qb[(size_t)(kc & 7) * HWp + ih * Wp + i0 + m];
    a[e] = (kc < 8) ? (_Float16)va : (_Float16)0.f;
    int kc2 = hh * 16 + e;
    float vb = kb[(size_t)(kc2 & 7) * HWp + ih * Wp + j0 + col];
    bb[e] = (kc2 < 8) ? (_Float16)vb : (_Float16)0.f;
  }
  v8f c = {};
  c = wmma16(a, bb, c);
#pragma unroll
  for (int r = 0; r < 8; r++)
    att[((size_t)(ih * Wp + i0 + hh * 8 + r)) * 384 + 192 + j0 + col] = c[r];
}

__global__ void k_softmax(float* __restrict__ att) {
  int row = blockIdx.x * 8 + (threadIdx.x >> 5);
  int lane = threadIdx.x & 31;
  float* p = att + (size_t)row * 384;
  float v[12]; float mx = -3.0e38f;
#pragma unroll
  for (int t = 0; t < 12; t++) { v[t] = p[t * 32 + lane]; mx = fmaxf(mx, v[t]); }
  mx = wred_max(mx);
  float s = 0.f;
#pragma unroll
  for (int t = 0; t < 12; t++) { v[t] = __expf(v[t] - mx); s += v[t]; }
  s = wred_sum(s);
  float inv = 1.f / s;
#pragma unroll
  for (int t = 0; t < 12; t++) p[t * 32 + lane] = v[t] * inv;
}

// outH: O[b,c,ih,iw] = sum_j v[b,c,j,iw] * attH[ih,iw,j]
__global__ void k_outH(const float* __restrict__ v, const float* __restrict__ att,
                       float* __restrict__ O, int b) {
  int gw = blockIdx.x * 4 + (threadIdx.x >> 5);
  int lane = threadIdx.x & 31;
  int iw = gw / 12, ih0 = (gw % 12) * 16;
  int col = lane & 15, hh = lane >> 4, m = col;
  const float* vb = v + (size_t)b * 64 * HWp;
  v8f acc[4] = {};
  for (int k0 = 0; k0 < 192; k0 += 32) {
    v16h bb;
#pragma unroll
    for (int e = 0; e < 16; e++)
      bb[e] = (_Float16)att[((size_t)((ih0 + col) * Wp + iw)) * 384 + k0 + hh * 16 + e];
#pragma unroll
    for (int mt = 0; mt < 4; mt++) {
      v16h a;
#pragma unroll
      for (int e = 0; e < 16; e++) {
        int kc = k0 + kA16(e, hh);
        a[e] = (_Float16)vb[(size_t)(mt * 16 + m) * HWp + kc * Wp + iw];
      }
      acc[mt] = wmma16(a, bb, acc[mt]);
    }
  }
#pragma unroll
  for (int mt = 0; mt < 4; mt++)
#pragma unroll
    for (int r = 0; r < 8; r++)
      O[((size_t)b * 64 + mt * 16 + hh * 8 + r) * HWp + (ih0 + col) * Wp + iw] = acc[mt][r];
}

// outW: O[b,c,ih,iw] += sum_j v[b,c,ih,j] * attW[ih,iw,j]
__global__ void k_outW(const float* __restrict__ v, const float* __restrict__ att,
                       float* __restrict__ O, int b) {
  int gw = blockIdx.x * 4 + (threadIdx.x >> 5);
  int lane = threadIdx.x & 31;
  int ih = gw / 12, iw0 = (gw % 12) * 16;
  int col = lane & 15, hh = lane >> 4, m = col;
  const float* vb = v + (size_t)b * 64 * HWp;
  v8f acc[4] = {};
  for (int k0 = 0; k0 < 192; k0 += 32) {
    v16h bb;
#pragma unroll
    for (int e = 0; e < 16; e++)
      bb[e] = (_Float16)att[((size_t)(ih * Wp + iw0 + col)) * 384 + 192 + k0 + hh * 16 + e];
#pragma unroll
    for (int mt = 0; mt < 4; mt++) {
      v16h a;
#pragma unroll
      for (int e = 0; e < 16; e++) {
        int kc = k0 + kA16(e, hh);
        a[e] = (_Float16)vb[(size_t)(mt * 16 + m) * HWp + ih * Wp + kc];
      }
      acc[mt] = wmma16(a, bb, acc[mt]);
    }
  }
#pragma unroll
  for (int mt = 0; mt < 4; mt++)
#pragma unroll
    for (int r = 0; r < 8; r++)
      O[((size_t)b * 64 + mt * 16 + hh * 8 + r) * HWp + ih * Wp + iw0 + col] += acc[mt][r];
}

__global__ void k_comb_y(float* __restrict__ O, const float* __restrict__ S,
                         const float* __restrict__ gamma) {
  int i = blockIdx.x * 256 + threadIdx.x;
  O[i] = S[i] + gamma[0] * O[i];
}

__global__ void k_lnpart(const float* __restrict__ O, float2* __restrict__ part) {
  int b = blockIdx.x >> 6, seg = blockIdx.x & 63, tid = threadIdx.x;
  const float* p = O + (size_t)b * 64 * HWp + (size_t)seg * 36864;
  float s = 0.f, s2 = 0.f;
  for (int i = tid; i < 36864; i += 256) { float v = p[i]; s += v; s2 += v * v; }
  __shared__ float sh[256], sh2[256];
  sh[tid] = s; sh2[tid] = s2; __syncthreads();
  for (int o = 128; o; o >>= 1) {
    if (tid < o) { sh[tid] += sh[tid + o]; sh2[tid] += sh2[tid + o]; }
    __syncthreads();
  }
  if (tid == 0) part[blockIdx.x] = make_float2(sh[0], sh2[0]);
}

__global__ void k_lnfin(const float2* __restrict__ part, float2* __restrict__ lnst) {
  int b = threadIdx.x;
  if (b >= 4) return;
  float s = 0.f, s2 = 0.f;
  for (int i = 0; i < 64; i++) { float2 p = part[b * 64 + i]; s += p.x; s2 += p.y; }
  float n = 64.0f * HWp, mu = s / n, var = s2 / n - mu * mu;
  lnst[b] = make_float2(mu, rsqrtf(var + EPSf));
}

__global__ void k_accln(float* __restrict__ acc, const float* __restrict__ O,
                        const float2* __restrict__ lnst, const float* __restrict__ f, int add) {
  int i = blockIdx.x * 256 + threadIdx.x;
  int b = i / (64 * HWp);
  float2 st = lnst[b];
  float v = f[0] * ((O[i] - st.x) * st.y);
  acc[i] = add ? (acc[i] + v) : v;
}

// ---------------- CAM ----------------
// K-split Gram: 16 splits of 72 K-blocks each -> partials, then deterministic reduce
__global__ void k_gram(const float* __restrict__ f, float* __restrict__ ep) {
  int gw = blockIdx.x;                       // 1024 waves, blockDim 32
  int lane = threadIdx.x;
  int ks = gw & 15, gid = gw >> 4;           // gid: 0..63
  int b = gid >> 4, mt = (gid >> 2) & 3, nt = gid & 3;
  int col = lane & 15, hh = lane >> 4, m = col;
  const float* fb_ = f + (size_t)b * 64 * HWp;
  int kbeg = ks * (HWp / 16);                // 2304 per split
  v8f acc = {};
  for (int k0 = kbeg; k0 < kbeg + HWp / 16; k0 += 32) {
    v16h a, bb;
#pragma unroll
    for (int e = 0; e < 16; e++) {
      a[e]  = (_Float16)fb_[(size_t)(mt * 16 + m) * HWp + k0 + kA16(e, hh)];
      bb[e] = (_Float16)fb_[(size_t)(nt * 16 + col) * HWp + k0 + hh * 16 + e];
    }
    acc = wmma16(a, bb, acc);
  }
#pragma unroll
  for (int r = 0; r < 8; r++)
    ep[(size_t)ks * 16384 + (size_t)b * 4096 + (mt * 16 + hh * 8 + r) * 64 + nt * 16 + col] = acc[r];
}

__global__ void k_gramred(const float* __restrict__ ep, float* __restrict__ e_) {
  int i = blockIdx.x * 256 + threadIdx.x;    // 16384 exact
  float s = 0.f;
#pragma unroll
  for (int ks = 0; ks < 16; ks++) s += ep[(size_t)ks * 16384 + i];
  e_[i] = s;
}

__global__ void k_camsm(const float* __restrict__ e_, _Float16* __restrict__ atth) {
  int row = blockIdx.x * 8 + (threadIdx.x >> 5);   // 256 rows
  int lane = threadIdx.x & 31;
  const float* p = e_ + (size_t)row * 64;
  float v0 = p[lane], v1 = p[lane + 32];
  float mx = wred_max(fmaxf(v0, v1));
  float t0 = mx - v0, t1 = mx - v1;
  float tm = wred_max(fmaxf(t0, t1));
  float e0 = __expf(t0 - tm), e1 = __expf(t1 - tm);
  float s = wred_sum(e0 + e1);
  float inv = 1.f / s;
  atth[(size_t)row * 64 + lane] = (_Float16)(e0 * inv);
  atth[(size_t)row * 64 + lane + 32] = (_Float16)(e1 * inv);
}

__global__ void k_pre(const float* __restrict__ x, const float* __restrict__ acc,
                      const float* __restrict__ camo, const float* __restrict__ cg,
                      float* __restrict__ pre) {
  int i = blockIdx.x * 256 + threadIdx.x;
  pre[i] = 2.f * x[i] + acc[i] + cg[0] * camo[i];
}

// ---------------- CBAM ----------------
__global__ void k_chstats(const float* __restrict__ pre, float2* __restrict__ st) {
  int tid = threadIdx.x;
  const float* p = pre + (size_t)blockIdx.x * HWp;
  float s = 0.f, mx = -3.0e38f;
  for (int i = tid; i < HWp; i += 256) { float v = p[i]; s += v; mx = fmaxf(mx, v); }
  __shared__ float sh[256], shm[256];
  sh[tid] = s; shm[tid] = mx; __syncthreads();
  for (int o = 128; o; o >>= 1) {
    if (tid < o) { sh[tid] += sh[tid + o]; shm[tid] = fmaxf(shm[tid], shm[tid + o]); }
    __syncthreads();
  }
  if (tid == 0) st[blockIdx.x] = make_float2(sh[0] * (1.0f / (float)HWp), shm[0]);
}

__global__ void k_cbammlp(const float2* __restrict__ st, const float* __restrict__ w1,
                          const float* __restrict__ w2, float* __restrict__ scale) {
  int t = threadIdx.x;
  int b = t >> 6, c = t & 63;
  float ha[4], hm[4];
  for (int j = 0; j < 4; j++) {
    float a = 0.f, mmx = 0.f;
    for (int cc = 0; cc < 64; cc++) {
      float2 s = st[b * 64 + cc];
      a += s.x * w1[j * 64 + cc];
      mmx += s.y * w1[j * 64 + cc];
    }
    ha[j] = fmaxf(a, 0.f); hm[j] = fmaxf(mmx, 0.f);
  }
  float o = 0.f;
  for (int j = 0; j < 4; j++) o += (ha[j] + hm[j]) * w2[c * 4 + j];
  scale[t] = sigmoidf(o);
}

__global__ void k_spstats(const float* __restrict__ pre, const float* __restrict__ scale,
                          float* __restrict__ sp) {
  int i = blockIdx.x * 256 + threadIdx.x;    // 147456 exact
  int b = i / HWp, p = i % HWp;
  float s = 0.f, mx = -3.0e38f;
  for (int c = 0; c < 64; c++) {
    float v = pre[((size_t)b * 64 + c) * HWp + p] * scale[b * 64 + c];
    s += v; mx = fmaxf(mx, v);
  }
  sp[((size_t)b * 2 + 0) * HWp + p] = s * (1.0f / 64.0f);
  sp[((size_t)b * 2 + 1) * HWp + p] = mx;
}

__global__ void k_saconv(const float* __restrict__ sp, const float* __restrict__ w,
                         float* __restrict__ sa) {
  int i = blockIdx.x * 256 + threadIdx.x;    // 147456 exact
  int b = i / HWp, p = i % HWp, y = p / Wp, x = p % Wp;
  float acc = 0.f;
  for (int ci = 0; ci < 2; ci++) {
    const float* s = sp + ((size_t)b * 2 + ci) * HWp;
    for (int ky = 0; ky < 7; ky++) {
      int yy = y + ky - 3; if (yy < 0 || yy >= Hp) continue;
      for (int kx = 0; kx < 7; kx++) {
        int xx = x + kx - 3; if (xx < 0 || xx >= Wp) continue;
        acc += s[yy * Wp + xx] * w[ci * 49 + ky * 7 + kx];
      }
    }
  }
  sa[i] = sigmoidf(acc);
}

__global__ void k_final(const float* __restrict__ pre, const float* __restrict__ scale,
                        const float* __restrict__ sa, float* __restrict__ out) {
  int i = blockIdx.x * 256 + threadIdx.x;
  int bc = i / HWp, b = bc >> 6, p = i % HWp;
  out[i] = pre[i] * scale[bc] * sa[(size_t)b * HWp + p];
}

// ---------------- host orchestration ----------------
extern "C" void kernel_launch(void* const* d_in, const int* in_sizes, int n_in,
                              void* d_out, int out_size, void* d_ws, size_t ws_size,
                              hipStream_t stream) {
  (void)in_sizes; (void)n_in; (void)out_size; (void)ws_size;
  const float* x     = (const float*)d_in[0];
  const float* fb    = (const float*)d_in[1];
  const float* oc_w  = (const float*)d_in[2];
  const float* oc_g  = (const float*)d_in[4];
  const float* oc_bt = (const float*)d_in[5];
  const float* pam_g  = (const float*)d_in[36];
  const float* pam_g2 = (const float*)d_in[37];
  const float* cam_g  = (const float*)d_in[38];
  const float* ca_w1  = (const float*)d_in[39];
  const float* ca_w2  = (const float*)d_in[40];
  const float* sa_w   = (const float*)d_in[41];

  float* ws = (float*)d_ws;
  const size_t IMG = (size_t)4 * 64 * HWp;       // 9437184
  float* B0  = ws;                                // pooled fb, later CAM output
  float* FBP = ws + IMG;
  float* S   = ws + 2 * IMG;
  float* V   = ws + 3 * IMG;
  float* O   = ws + 4 * IMG;
  float* ACC = ws + 5 * IMG;
  float* PRE = ws + 6 * IMG;
  float* Q   = ws + 7 * IMG;                      // 4*8*HWp = 1179648
  float* Kb  = Q + 1179648;
  float* ATT = Kb + 1179648;                      // per-batch: 36864*384 = 14155776
  float* SM  = ATT + 14155776;
  float*  seb  = SM;                              // 256
  float*  scl  = SM + 256;                        // 256
  float2* bnst = (float2*)(SM + 512);             // 64 float2
  float2* part = (float2*)(SM + 640);             // 256 float2
  float2* lnst = (float2*)(SM + 1152);            // 4 float2
  float*  ECAM = SM + 1160;                       // 4*4096
  float2* chst = (float2*)(SM + 17544);           // 256 float2
  float*  SP   = SM + 18056;                      // 4*2*HWp
  float*  SA   = SP + 294912;                     // 4*HWp
  _Float16* HREG   = (_Float16*)(SA + 147456);    // 32768 halfs = 16384 floats
  _Float16* WH_OC  = HREG;
  _Float16* WH_Q0  = HREG + 4096;
  _Float16* WH_K0  = HREG + 5120;
  _Float16* WH_V0  = HREG + 6144;
  _Float16* WH_Q1  = HREG + 10240;
  _Float16* WH_K1  = HREG + 11264;
  _Float16* WH_V1  = HREG + 12288;
  _Float16* CATT_H = HREG + 16384;
  float* GPART = SA + 147456 + 16384;             // 16*16384 = 262144 floats

  // fb pipeline: maxpool -> conv1x1 -> BN+relu  (conv bias cancels under BN)
  k_maxpool2<<<36864, 256, 0, stream>>>(fb, B0);
  auto cvt = [&](const float* w, _Float16* dst, int rows, int padRows) {
    k_cvt_w<<<(padRows * 64) / 256, 256, 0, stream>>>(w, dst, rows, 64);
  };
  cvt(oc_w, WH_OC, 64, 64);
  cvt((const float*)d_in[6 + 2],  WH_Q0, 8, 16);
  cvt((const float*)d_in[6 + 6],  WH_K0, 8, 16);
  cvt((const float*)d_in[6 + 10], WH_V0, 64, 64);
  cvt((const float*)d_in[21 + 2],  WH_Q1, 8, 16);
  cvt((const float*)d_in[21 + 6],  WH_K1, 8, 16);
  cvt((const float*)d_in[21 + 10], WH_V1, 64, 64);

  k_conv1x1<64, 4><<<2304, 128, 0, stream>>>(B0, FBP, WH_OC, 0);
  k_bnstats<<<64, 256, 0, stream>>>(FBP, bnst, 64);
  k_bnapply<<<36864, 256, 0, stream>>>(FBP, bnst, oc_g, oc_bt, 64, 1);

  auto smodule = [&](const float* X, int pbase, _Float16* WHq, _Float16* WHk, _Float16* WHv,
                     const float* fscale, int add) {
    const float* P[15];
    for (int i = 0; i < 15; i++) P[i] = (const float*)d_in[pbase + i];
    k_semean<<<256, 256, 0, stream>>>(X, seb);
    k_semlp<<<1, 256, 0, stream>>>(seb, P[0], P[1], scl);
    k_seapply<<<36864, 256, 0, stream>>>(X, scl, S);
    // q = relu(bn(conv(S)))  (8 channels, padded to 16)
    k_conv1x1<8, 1><<<2304, 128, 0, stream>>>(S, Q, WHq, 0);
    k_bnstats<<<8, 256, 0, stream>>>(Q, bnst, 8);
    k_bnapply<<<4608, 256, 0, stream>>>(Q, bnst, P[4], P[5], 8, 1);
    // k
    k_conv1x1<8, 1><<<2304, 128, 0, stream>>>(S, Kb, WHk, 0);
    k_bnstats<<<8, 256, 0, stream>>>(Kb, bnst, 8);
    k_bnapply<<<4608, 256, 0, stream>>>(Kb, bnst, P[8], P[9], 8, 1);
    // v = bn(conv(S))
    k_conv1x1<64, 4><<<2304, 128, 0, stream>>>(S, V, WHv, 0);
    k_bnstats<<<64, 256, 0, stream>>>(V, bnst, 64);
    k_bnapply<<<36864, 256, 0, stream>>>(V, bnst, P[12], P[13], 64, 0);
    // criss-cross attention, one batch at a time (ATT scratch = 56.6MB)
    for (int b = 0; b < 4; b++) {
      k_eH<<<6912, 128, 0, stream>>>(Q, Kb, ATT, b);
      k_eW<<<6912, 128, 0, stream>>>(Q, Kb, ATT, b);
      k_softmax<<<4608, 256, 0, stream>>>(ATT);
      k_outH<<<576, 128, 0, stream>>>(V, ATT, O, b);
      k_outW<<<576, 128, 0, stream>>>(V, ATT, O, b);
    }
    k_comb_y<<<36864, 256, 0, stream>>>(O, S, P[14]);          // y = S + gamma*(outH+outW)
    k_lnpart<<<256, 256, 0, stream>>>(O, part);
    k_lnfin<<<1, 32, 0, stream>>>(part, lnst);
    k_accln<<<36864, 256, 0, stream>>>(ACC, O, lnst, fscale, add);
  };
  smodule(x,   6,  WH_Q0, WH_K0, WH_V0, pam_g,  0);
  smodule(FBP, 21, WH_Q1, WH_K1, WH_V1, pam_g2, 1);

  // CAM: K-split gram -> reduce -> (max-e) softmax -> apply; PRE = 2x + ACC + cam_g*cam_out
  k_gram<<<1024, 32, 0, stream>>>(FBP, GPART);
  k_gramred<<<64, 256, 0, stream>>>(GPART, ECAM);
  k_camsm<<<32, 256, 0, stream>>>(ECAM, CATT_H);
  k_conv1x1<64, 4><<<2304, 128, 0, stream>>>(FBP, B0, CATT_H, 4096);
  k_pre<<<36864, 256, 0, stream>>>(x, ACC, B0, cam_g, PRE);

  // CBAM
  k_chstats<<<256, 256, 0, stream>>>(PRE, chst);
  k_cbammlp<<<1, 256, 0, stream>>>(chst, ca_w1, ca_w2, scl);
  k_spstats<<<576, 256, 0, stream>>>(PRE, scl, SP);
  k_saconv<<<576, 256, 0, stream>>>(SP, sa_w, SA);
  k_final<<<36864, 256, 0, stream>>>(PRE, scl, SA, (float*)d_out);
}
